// Qwen3MoE_56650618634249
// MI455X (gfx1250) — matrix operand order repaired and checked by reading, not compiled, not run on hardware
//
#include <hip/hip_runtime.h>
#include <hip/hip_bf16.h>

// ---------------------------------------------------------------------------
// Qwen3-style MoE for MI455X (gfx1250).
// A = weights (row-major, direct global b128 fragment loads),
// B = activations stored TRANSPOSED [k][token] (contiguous fragment rows),
// 64-token column tiles: each A fragment feeds 4 WMMAs (8/iter in ffn1).
// bf16 WMMA, f32 accumulate, double-buffered LDS, 1 barrier per k-step.
// ---------------------------------------------------------------------------

typedef __attribute__((ext_vector_type(16))) __bf16 v16bf;
typedef __attribute__((ext_vector_type(8)))  float  v8f;

struct U32x8 { uint4 a; uint4 b; };

#define E_   16
#define D_   1024
#define H_   1024
#define SH_  2048
#define NT   64      // token-column tile
#define XSTR 72      // LDS B-tile row stride (halfwords): 144B, 16B-aligned + padded

__device__ __forceinline__ unsigned short f2bf(float f) {
  unsigned u = __builtin_bit_cast(unsigned, f);
  u += 0x7FFFu + ((u >> 16) & 1u);
  return (unsigned short)(u >> 16);
}

__device__ __forceinline__ v8f wmma_bf16(v16bf a, v16bf b, v8f c) {
  return __builtin_amdgcn_wmma_f32_16x16x32_bf16(false, a, false, b, (short)0, c,
                                                 false, false);
}

// A fragment (16x32 bf16) straight from global row-major weights.
__device__ __forceinline__ v16bf fragAg(const unsigned short* Wrow, long ldw,
                                        int lane, int k0) {
  int m = lane & 15, kh = lane >> 4;
  const unsigned short* r = Wrow + (long)m * ldw + k0 + kh * 8;
  U32x8 u; u.a = *(const uint4*)r; u.b = *(const uint4*)(r + 16);
  return __builtin_bit_cast(v16bf, u);
}

// B fragment (32x16 bf16): lane column n=lane%16, elements k = 8h+s / 16+8h+(s-8) (table order).
__device__ __forceinline__ v16bf fragBs(const unsigned short* sX, int lane, int n0) {
  int n = lane & 15, kh = lane >> 4;
  const unsigned short* r = sX + n0 + n;
  unsigned short v[16];
#pragma unroll
  for (int s = 0; s < 8; ++s) {
    v[s] = r[(kh * 8 + s) * XSTR];
    v[8 + s] = r[(16 + kh * 8 + s) * XSTR];
  }
  return __builtin_bit_cast(v16bf, v);
}

// Stage a 32(k) x 64(col) bf16 tile from a transposed activation buffer
// (row-major [k][col], stride ld) into LDS: 1 uint4 load + 1 b128 store/thread.
__device__ __forceinline__ void stageX(unsigned short* dst, const unsigned short* src,
                                       long ld, long c0, int k0, int t) {
  int r = t >> 3, c = (t & 7) * 8;
  *(uint4*)&dst[r * XSTR + c] = *(const uint4*)(src + (long)(k0 + r) * ld + c0 + c);
}

// ---------------------------------------------------------------------------
// K0a: f32 -> bf16 (weights, row-major preserved)
// ---------------------------------------------------------------------------
__global__ void k_cvt(const float* __restrict__ src, unsigned short* __restrict__ dst,
                      long n) {
  long i = ((long)blockIdx.x * blockDim.x + threadIdx.x) * 8;
  if (i >= n) return;
  float4 a = *(const float4*)(src + i);
  float4 b = *(const float4*)(src + i + 4);
  unsigned short o[8] = { f2bf(a.x), f2bf(a.y), f2bf(a.z), f2bf(a.w),
                          f2bf(b.x), f2bf(b.y), f2bf(b.z), f2bf(b.w) };
  *(uint4*)(dst + i) = *(const uint4*)o;
}

// K0b: x [ntok][D] f32 -> xbT [D][ntok] bf16 (tiled transpose through LDS)
__global__ void k_cvt_xT(const float* __restrict__ x, unsigned short* __restrict__ xbT,
                         int ntok) {
  __shared__ unsigned short tile[32][40];
  int t = threadIdx.x;
  int t0 = blockIdx.x * 32, d0 = blockIdx.y * 32;
  {
    int tr = t >> 3, dc = (t & 7) * 4;
    float4 v = *(const float4*)(x + (long)(t0 + tr) * D_ + d0 + dc);
    tile[tr][dc + 0] = f2bf(v.x); tile[tr][dc + 1] = f2bf(v.y);
    tile[tr][dc + 2] = f2bf(v.z); tile[tr][dc + 3] = f2bf(v.w);
  }
  __syncthreads();
  {
    int dr = t >> 3, tc = (t & 7) * 4;
    unsigned short o[4] = { tile[tc + 0][dr], tile[tc + 1][dr],
                            tile[tc + 2][dr], tile[tc + 3][dr] };
    *(uint2*)(xbT + (long)(d0 + dr) * ntok + t0 + tc) = *(const uint2*)o;
  }
}

// ---------------------------------------------------------------------------
// K1: router
// ---------------------------------------------------------------------------
__global__ void k_router(const float* __restrict__ x, const float* __restrict__ gw,
                         const float* __restrict__ sgw,
                         int* __restrict__ top_i, float* __restrict__ top_w,
                         float* __restrict__ sg, int* __restrict__ counts, int ntok) {
  int n = blockIdx.x * blockDim.x + threadIdx.x;
  if (n >= ntok) return;
  const float* xr = x + (long)n * D_;
  float lg[E_];
#pragma unroll
  for (int e = 0; e < E_; ++e) lg[e] = 0.f;
  float sga = 0.f;
  for (int d = 0; d < D_; ++d) {
    float xv = xr[d];
    sga += xv * sgw[d];
#pragma unroll
    for (int e = 0; e < E_; ++e) lg[e] += xv * gw[e * D_ + d];
  }
  int i1 = 0;
#pragma unroll
  for (int e = 1; e < E_; ++e) if (lg[e] > lg[i1]) i1 = e;
  int i2 = (i1 == 0) ? 1 : 0;
#pragma unroll
  for (int e = 0; e < E_; ++e) if (e != i1 && lg[e] > lg[i2]) i2 = e;
  float e2 = __expf(lg[i2] - lg[i1]);
  float inv = 1.f / (1.f + e2);
  top_i[2 * n] = i1; top_i[2 * n + 1] = i2;
  top_w[2 * n] = inv; top_w[2 * n + 1] = e2 * inv;
  sg[n] = 1.f / (1.f + __expf(-sga));
  atomicAdd(&counts[i1], 1);
  atomicAdd(&counts[i2], 1);
}

// K2: exclusive scan, offsets aligned to 64 so tile column bases stay aligned
__global__ void k_scan(const int* __restrict__ counts, int* __restrict__ offs,
                       int* __restrict__ cursors) {
  if (threadIdx.x == 0) {
    int acc = 0;
    for (int e = 0; e < E_; ++e) {
      offs[e] = acc; cursors[e] = acc;
      acc += (counts[e] + 63) & ~63;
    }
  }
}

// K3: fill per-expert assignment lists at padded positions
__global__ void k_fill(const int* __restrict__ top_i, const float* __restrict__ top_w,
                       int* __restrict__ cursors, int* __restrict__ tokP,
                       int* __restrict__ slotP, float* __restrict__ wtP, int na) {
  int i = blockIdx.x * blockDim.x + threadIdx.x;
  if (i >= na) return;
  int pos = atomicAdd(&cursors[top_i[i]], 1);
  tokP[pos] = i >> 1;
  slotP[pos] = i & 1;
  wtP[pos] = top_w[i];
}

// K4: column gather xgT[k][p] = xbT[k][tokP[p]]  (coalesced writes)
__global__ void k_gather(const unsigned short* __restrict__ xbT,
                         const int* __restrict__ tokP,
                         unsigned short* __restrict__ xgT, int ntok, int naP) {
  int p = blockIdx.x * 256 + threadIdx.x;
  if (p >= naP) return;
  long k = blockIdx.y;
  xgT[k * naP + p] = xbT[k * ntok + tokP[p]];
}

// ---------------------------------------------------------------------------
// K5: per-expert  hT = ((W1 @ xT) * silu(Wg @ xT))   -> hbufT [H][naP]
// grid (tilesA, H/128, E), block 256: 8 WMMAs per k-step per wave.
// ---------------------------------------------------------------------------
__global__ void __launch_bounds__(256)
k_ffn1(const unsigned short* __restrict__ xgT, const unsigned short* __restrict__ w1b,
       const unsigned short* __restrict__ wgb, const int* __restrict__ offs,
       const int* __restrict__ counts, unsigned short* __restrict__ hbufT, int naP) {
  int e = blockIdx.z;
  int cnt = counts[e];
  int a0 = blockIdx.x * NT;
  if (a0 >= cnt) return;
  long c0 = (long)offs[e] + a0;
  long cEnd = (long)offs[e] + cnt;
  int t = threadIdx.x, lane = t & 31;
  int mb = blockIdx.y * 128 + ((t >> 5) << 4);
  __shared__ __align__(16) unsigned short sX[2][32 * XSTR];
  const unsigned short* W1 = w1b + (long)e * H_ * D_ + (long)mb * D_;
  const unsigned short* Wg = wgb + (long)e * H_ * D_ + (long)mb * D_;
  v8f a1[4] = {{}, {}, {}, {}};
  v8f ag[4] = {{}, {}, {}, {}};
  stageX(sX[0], xgT, naP, c0, 0, t);
  __syncthreads();
  for (int k0 = 0; k0 < D_; k0 += 32) {
    int cur = (k0 >> 5) & 1;
    if (k0 + 32 < D_) stageX(sX[cur ^ 1], xgT, naP, c0, k0 + 32, t);
    v16bf fa1 = fragAg(W1, D_, lane, k0);
    v16bf fag = fragAg(Wg, D_, lane, k0);
#pragma unroll
    for (int s = 0; s < 4; ++s) {
      v16bf fb = fragBs(sX[cur], lane, s * 16);
      a1[s] = wmma_bf16(fa1, fb, a1[s]);
      ag[s] = wmma_bf16(fag, fb, ag[s]);
    }
    __syncthreads();
  }
  int kh = lane >> 4, nn = lane & 15;
#pragma unroll
  for (int s = 0; s < 4; ++s) {
    long cc = c0 + s * 16 + nn;
    if (cc < cEnd) {
#pragma unroll
      for (int r = 0; r < 8; ++r) {
        int h = mb + kh * 8 + r;
        float g = ag[s][r];
        hbufT[(long)h * naP + cc] = f2bf(a1[s][r] * (g / (1.f + __expf(-g))));
      }
    }
  }
}

// ---------------------------------------------------------------------------
// K6: per-expert  contribT[slot][d][tok] = w * (W2 @ hT)
// grid (tilesA, D/128, E)
// ---------------------------------------------------------------------------
__global__ void __launch_bounds__(256)
k_ffn2(const unsigned short* __restrict__ hbufT, const unsigned short* __restrict__ w2b,
       const int* __restrict__ tokP, const int* __restrict__ slotP,
       const float* __restrict__ wtP, const int* __restrict__ offs,
       const int* __restrict__ counts, float* __restrict__ contribT,
       int ntok, int naP) {
  int e = blockIdx.z;
  int cnt = counts[e];
  int a0 = blockIdx.x * NT;
  if (a0 >= cnt) return;
  long c0 = (long)offs[e] + a0;
  int nvalid = cnt - a0; if (nvalid > NT) nvalid = NT;
  int t = threadIdx.x, lane = t & 31;
  int mb = blockIdx.y * 128 + ((t >> 5) << 4);
  __shared__ __align__(16) unsigned short sX[2][32 * XSTR];
  __shared__ int sTok[NT]; __shared__ int sSlot[NT]; __shared__ float sWt[NT];
  if (t < NT) {
    long p = c0 + (t < nvalid ? t : nvalid - 1);
    sTok[t] = tokP[p]; sSlot[t] = slotP[p]; sWt[t] = wtP[p];
  }
  const unsigned short* W2 = w2b + (long)e * D_ * H_ + (long)mb * H_;
  v8f ac[4] = {{}, {}, {}, {}};
  stageX(sX[0], hbufT, naP, c0, 0, t);
  __syncthreads();
  for (int k0 = 0; k0 < H_; k0 += 32) {
    int cur = (k0 >> 5) & 1;
    if (k0 + 32 < H_) stageX(sX[cur ^ 1], hbufT, naP, c0, k0 + 32, t);
    v16bf fa = fragAg(W2, H_, lane, k0);
#pragma unroll
    for (int s = 0; s < 4; ++s) {
      v16bf fb = fragBs(sX[cur], lane, s * 16);
      ac[s] = wmma_bf16(fa, fb, ac[s]);
    }
    __syncthreads();
  }
  int kh = lane >> 4, nn = lane & 15;
#pragma unroll
  for (int s = 0; s < 4; ++s) {
    int nc = s * 16 + nn;
    if (nc < nvalid) {
      long base = (long)sSlot[nc] * D_ * ntok + sTok[nc];
      float wt = sWt[nc];
#pragma unroll
      for (int r = 0; r < 8; ++r) {
        int d = mb + kh * 8 + r;
        contribT[base + (long)d * ntok] = wt * ac[s][r];
      }
    }
  }
}

// ---------------------------------------------------------------------------
// K7: shared expert  shT = (W1s @ xT) * silu(Wgs @ xT)  -> shbufT [SH][ntok]
// grid (ntok/64, SH/128)
// ---------------------------------------------------------------------------
__global__ void __launch_bounds__(256)
k_shared1(const unsigned short* __restrict__ xbT, const unsigned short* __restrict__ w1s,
          const unsigned short* __restrict__ wgs, unsigned short* __restrict__ shbufT,
          int ntok) {
  long c0 = (long)blockIdx.x * NT;
  int t = threadIdx.x, lane = t & 31;
  int mb = blockIdx.y * 128 + ((t >> 5) << 4);
  __shared__ __align__(16) unsigned short sX[2][32 * XSTR];
  const unsigned short* W1 = w1s + (long)mb * D_;
  const unsigned short* Wg = wgs + (long)mb * D_;
  v8f a1[4] = {{}, {}, {}, {}};
  v8f ag[4] = {{}, {}, {}, {}};
  stageX(sX[0], xbT, ntok, c0, 0, t);
  __syncthreads();
  for (int k0 = 0; k0 < D_; k0 += 32) {
    int cur = (k0 >> 5) & 1;
    if (k0 + 32 < D_) stageX(sX[cur ^ 1], xbT, ntok, c0, k0 + 32, t);
    v16bf fa1 = fragAg(W1, D_, lane, k0);
    v16bf fag = fragAg(Wg, D_, lane, k0);
#pragma unroll
    for (int s = 0; s < 4; ++s) {
      v16bf fb = fragBs(sX[cur], lane, s * 16);
      a1[s] = wmma_bf16(fa1, fb, a1[s]);
      ag[s] = wmma_bf16(fag, fb, ag[s]);
    }
    __syncthreads();
  }
  int kh = lane >> 4, nn = lane & 15;
#pragma unroll
  for (int s = 0; s < 4; ++s) {
    long cc = c0 + s * 16 + nn;
#pragma unroll
    for (int r = 0; r < 8; ++r) {
      int h = mb + kh * 8 + r;
      float g = ag[s][r];
      shbufT[(long)h * ntok + cc] = f2bf(a1[s][r] * (g / (1.f + __expf(-g))));
    }
  }
}

// ---------------------------------------------------------------------------
// K8: outT[d][tok] = contrib0 + contrib1 + sg[tok] * (W2s @ shT)
// grid (ntok/64, D/128)
// ---------------------------------------------------------------------------
__global__ void __launch_bounds__(256)
k_final(const unsigned short* __restrict__ shbufT, const unsigned short* __restrict__ w2s,
        const float* __restrict__ contribT, const float* __restrict__ sg,
        float* __restrict__ outT, int ntok) {
  long c0 = (long)blockIdx.x * NT;
  int t = threadIdx.x, lane = t & 31;
  int mb = blockIdx.y * 128 + ((t >> 5) << 4);
  __shared__ __align__(16) unsigned short sX[2][32 * XSTR];
  const unsigned short* W2 = w2s + (long)mb * SH_;
  v8f ac[4] = {{}, {}, {}, {}};
  stageX(sX[0], shbufT, ntok, c0, 0, t);
  __syncthreads();
  for (int k0 = 0; k0 < SH_; k0 += 32) {
    int cur = (k0 >> 5) & 1;
    if (k0 + 32 < SH_) stageX(sX[cur ^ 1], shbufT, ntok, c0, k0 + 32, t);
    v16bf fa = fragAg(W2, SH_, lane, k0);
#pragma unroll
    for (int s = 0; s < 4; ++s) {
      v16bf fb = fragBs(sX[cur], lane, s * 16);
      ac[s] = wmma_bf16(fa, fb, ac[s]);
    }
    __syncthreads();
  }
  long nd = (long)D_ * ntok;
  int kh = lane >> 4, nn = lane & 15;
#pragma unroll
  for (int s = 0; s < 4; ++s) {
    long tok = c0 + s * 16 + nn;
    float sgv = sg[tok];
#pragma unroll
    for (int r = 0; r < 8; ++r) {
      int d = mb + kh * 8 + r;
      long i = (long)d * ntok + tok;
      outT[i] = contribT[i] + contribT[nd + i] + sgv * ac[s][r];
    }
  }
}

// K9: outT [D][ntok] -> out [ntok][D] (tiled f32 transpose)
__global__ void k_trans_out(const float* __restrict__ outT, float* __restrict__ out,
                            int ntok) {
  __shared__ float tile[32][33];
  int t = threadIdx.x;
  int t0 = blockIdx.x * 32, d0 = blockIdx.y * 32;
  {
    int dr = t >> 3, tc = (t & 7) * 4;
    float4 v = *(const float4*)(outT + (long)(d0 + dr) * ntok + t0 + tc);
    tile[dr][tc] = v.x; tile[dr][tc + 1] = v.y;
    tile[dr][tc + 2] = v.z; tile[dr][tc + 3] = v.w;
  }
  __syncthreads();
  {
    int tr = t >> 3, dc = (t & 7) * 4;
    float4 v = { tile[dc][tr], tile[dc + 1][tr], tile[dc + 2][tr], tile[dc + 3][tr] };
    *(float4*)(out + (long)(t0 + tr) * D_ + d0 + dc) = v;
  }
}

// ---------------------------------------------------------------------------
// Host launcher
// ---------------------------------------------------------------------------
extern "C" void kernel_launch(void* const* d_in, const int* in_sizes, int n_in,
                              void* d_out, int out_size, void* d_ws, size_t ws_size,
                              hipStream_t stream) {
  const float* x   = (const float*)d_in[0];
  const float* gw  = (const float*)d_in[1];
  const float* wg  = (const float*)d_in[2];
  const float* w1  = (const float*)d_in[3];
  const float* w2  = (const float*)d_in[4];
  const float* swg = (const float*)d_in[5];
  const float* sw1 = (const float*)d_in[6];
  const float* sw2 = (const float*)d_in[7];
  const float* sgw = (const float*)d_in[8];
  float* out = (float*)d_out;

  const int ntok = in_sizes[0] / D_;
  const int na   = ntok * 2;
  const int naP  = na + NT * E_;       // padded assignment columns (64-aligned offs)

  char* p = (char*)d_ws;
  auto alloc = [&](size_t bytes) { char* r = p; p += (bytes + 255) & ~(size_t)255; return r; };
  unsigned short* xbT   = (unsigned short*)alloc((size_t)D_ * ntok * 2);
  unsigned short* xgT   = (unsigned short*)alloc((size_t)D_ * naP * 2);
  unsigned short* wgb   = (unsigned short*)alloc((size_t)E_ * H_ * D_ * 2);
  unsigned short* w1b   = (unsigned short*)alloc((size_t)E_ * H_ * D_ * 2);
  unsigned short* w2b   = (unsigned short*)alloc((size_t)E_ * D_ * H_ * 2);
  unsigned short* swgb  = (unsigned short*)alloc((size_t)SH_ * D_ * 2);
  unsigned short* sw1b  = (unsigned short*)alloc((size_t)SH_ * D_ * 2);
  unsigned short* sw2b  = (unsigned short*)alloc((size_t)D_ * SH_ * 2);
  unsigned short* hbufT = (unsigned short*)alloc((size_t)H_ * naP * 2);
  unsigned short* shbufT= (unsigned short*)alloc((size_t)SH_ * ntok * 2);
  float* contribT       = (float*)alloc((size_t)2 * D_ * ntok * 4);
  float* outT           = (float*)alloc((size_t)D_ * ntok * 4);
  int*   top_i          = (int*)alloc((size_t)na * 4);
  float* top_w          = (float*)alloc((size_t)na * 4);
  float* sg             = (float*)alloc((size_t)ntok * 4);
  int*   counts         = (int*)alloc(64);
  int*   offs           = (int*)alloc(64);
  int*   cursors        = (int*)alloc(64);
  int*   tokP           = (int*)alloc((size_t)naP * 4);
  int*   slotP          = (int*)alloc((size_t)naP * 4);
  float* wtP            = (float*)alloc((size_t)naP * 4);

  auto cvt = [&](const float* s, unsigned short* d, long n) {
    long thr = n / 8;
    k_cvt<<<dim3((unsigned)((thr + 255) / 256)), dim3(256), 0, stream>>>(s, d, n);
  };
  cvt(wg,  wgb,  (long)E_ * H_ * D_);
  cvt(w1,  w1b,  (long)E_ * H_ * D_);
  cvt(w2,  w2b,  (long)E_ * D_ * H_);
  cvt(swg, swgb, (long)SH_ * D_);
  cvt(sw1, sw1b, (long)SH_ * D_);
  cvt(sw2, sw2b, (long)D_ * SH_);
  k_cvt_xT<<<dim3(ntok / 32, D_ / 32), dim3(256), 0, stream>>>(x, xbT, ntok);

  hipMemsetAsync(counts, 0, E_ * sizeof(int), stream);
  hipMemsetAsync(tokP, 0, (size_t)naP * sizeof(int), stream);
  k_router<<<dim3((ntok + 127) / 128), dim3(128), 0, stream>>>(
      x, gw, sgw, top_i, top_w, sg, counts, ntok);
  k_scan<<<dim3(1), dim3(32), 0, stream>>>(counts, offs, cursors);
  k_fill<<<dim3((na + 255) / 256), dim3(256), 0, stream>>>(
      top_i, top_w, cursors, tokP, slotP, wtP, na);
  k_gather<<<dim3((naP + 255) / 256, D_), dim3(256), 0, stream>>>(
      xbT, tokP, xgT, ntok, naP);

  int tilesA = (ntok + NT - 1) / NT;   // worst-case tiles per expert
  k_ffn1<<<dim3(tilesA, H_ / 128, E_), dim3(256), 0, stream>>>(
      xgT, w1b, wgb, offs, counts, hbufT, naP);
  k_ffn2<<<dim3(tilesA, D_ / 128, E_), dim3(256), 0, stream>>>(
      hbufT, w2b, tokP, slotP, wtP, offs, counts, contribT, ntok, naP);
  k_shared1<<<dim3(ntok / NT, SH_ / 128), dim3(256), 0, stream>>>(
      xbT, sw1b, swgb, shbufT, ntok);
  k_final<<<dim3(ntok / NT, D_ / 128), dim3(256), 0, stream>>>(
      shbufT, sw2b, contribT, sg, outT, ntok);
  k_trans_out<<<dim3(ntok / 32, D_ / 32), dim3(256), 0, stream>>>(outT, out, ntok);
}
